// SimpleSAE_46059229282443
// MI455X (gfx1250) — matrix operand order repaired and checked by reading, not compiled, not run on hardware
//
#include <hip/hip_runtime.h>
#include <hip/hip_bf16.h>
#include <stdint.h>

typedef __attribute__((ext_vector_type(16))) __bf16 v16bf;
typedef __attribute__((ext_vector_type(8)))  float  v8f;

#define IN_DIM 768
#define HID    1024
#define KTOP   50
#define KT1 (IN_DIM / 32)   // 24 k-tiles for encoder GEMM
#define KT2 (HID / 32)      // 32 k-tiles for decoder GEMM
#define ROWS_WG 32          // rows per workgroup (2 row-tiles of 16)
#define HSTR   1028         // padded h-tile row stride (floats): 8*1028 % 64 = 32 -> no bank clash

__device__ __forceinline__ unsigned short f2bf(float f) {
    unsigned u = __float_as_uint(f);
    u += 0x7FFFu + ((u >> 16) & 1u);   // round-to-nearest-even
    return (unsigned short)(u >> 16);
}

__device__ __forceinline__ float fast_tanh(float x) {
    // tanh(x) = 1 - 2/(exp(2x)+1); v_exp_f32 + raw v_rcp_f32, saturates to +-1.
    float e = __expf(2.0f * x);
    return 1.0f - 2.0f * __builtin_amdgcn_rcpf(e + 1.0f);
}

union FragU { v16bf v; uint4 q[2]; };

__device__ __forceinline__ v16bf load_frag(const unsigned short* p) {
    FragU f;
    const uint4* q = (const uint4*)p;   // 32B-aligned by construction
    f.q[0] = q[0];
    f.q[1] = q[1];
    return f.v;
}

// Map (row-in-tile m, k-in-tile kk) -> (lane, element) per CDNA5 16-bit
// A-matrix 16x32 VGPR layout (ISA 7.12.2).
__device__ __forceinline__ void a_lane_elem(int m, int kk, int& l, int& e) {
    if (kk < 8)       { l = m;      e = kk;      }
    else if (kk < 16) { l = m + 16; e = kk - 8;  }
    else if (kk < 24) { l = m;      e = kk - 8;  }
    else              { l = m + 16; e = kk - 16; }
}

// Convert fp32 weight matrix W[K x ldn] into bf16 WMMA B-fragments:
// frag (nt, kt) holds the 32x16 tile, 512 bf16 contiguous, per-lane 16
// contiguous bf16 (lane<16: K=0..7,16..23, lane>=16: K=8..15,24..31, col = lane&15).
__global__ void __launch_bounds__(256) prep_w(const float* __restrict__ W,
                                              unsigned short* __restrict__ out,
                                              int ktiles, int ldn, int total) {
    int d = blockIdx.x * blockDim.x + threadIdx.x;
    if (d >= total) return;
    int frag = d >> 9;           // /512
    int r    = d & 511;
    int lane = r >> 4;
    int e    = r & 15;
    int nt   = frag / ktiles;
    int kt   = frag - nt * ktiles;
    int col  = lane & 15;
    int kk   = (e < 8) ? ((lane >> 4) * 8 + e) : (16 + (lane >> 4) * 8 + (e - 8));
    out[d] = f2bf(W[(size_t)(kt * 32 + kk) * ldn + nt * 16 + col]);
}

// Encoder: x[32 x 768] @ W1 -> +b1 -> LayerNorm -> ReLU -> top-K mask -> code
__global__ void __launch_bounds__(256) sae_encoder(const float* __restrict__ x,
                                                   const unsigned short* __restrict__ w1f,
                                                   const float* __restrict__ b1,
                                                   const float* __restrict__ gamma,
                                                   const float* __restrict__ beta,
                                                   float* __restrict__ code) {
    extern __shared__ char smem[];
    unsigned short* aF = (unsigned short*)smem;                              // 2*KT1*512 bf16 (48 KB)
    float* hT = (float*)(smem + 2 * KT1 * 512 * sizeof(unsigned short));     // 16*HSTR f32 (~64 KB)

    const int tid  = threadIdx.x;
    const int lane = tid & 31;
    const int wave = tid >> 5;
    const int row0 = blockIdx.x * ROWS_WG;

    // Stage both 16-row x tiles as bf16 A-fragments (coalesced global reads).
    for (int idx = tid; idx < ROWS_WG * IN_DIM; idx += 256) {
        int m = idx / IN_DIM;            // 0..31
        int k = idx - m * IN_DIM;
        float v = x[(size_t)(row0 + m) * IN_DIM + k];
        int tile = m >> 4, mm = m & 15;
        int kt = k >> 5, kk = k & 31, l, e;
        a_lane_elem(mm, kk, l, e);
        aF[tile * (KT1 * 512) + kt * 512 + l * 16 + e] = f2bf(v);
    }
    __syncthreads();

    v8f acc[2][8];
    #pragma unroll
    for (int rt = 0; rt < 2; ++rt)
        #pragma unroll
        for (int t = 0; t < 8; ++t)
            #pragma unroll
            for (int v = 0; v < 8; ++v) acc[rt][t][v] = 0.f;

    const int ntBase = wave * 8;           // each wave owns 128 of 1024 cols
    for (int kt = 0; kt < KT1; ++kt) {
        v16bf a0 = load_frag(aF + kt * 512 + lane * 16);
        v16bf a1 = load_frag(aF + KT1 * 512 + kt * 512 + lane * 16);
        #pragma unroll
        for (int t = 0; t < 8; ++t) {
            v16bf b = load_frag(w1f + ((size_t)(ntBase + t) * KT1 + kt) * 512 + lane * 16);
            acc[0][t] = __builtin_amdgcn_wmma_f32_16x16x32_bf16(
                false, a0, false, b, (short)0, acc[0][t], false, false);
            acc[1][t] = __builtin_amdgcn_wmma_f32_16x16x32_bf16(
                false, a1, false, b, (short)0, acc[1][t], false, false);
        }
    }

    // Epilogue: two phases over the shared 16-row h tile.
    const int wc0 = wave * 128;
    for (int half = 0; half < 2; ++half) {
        __syncthreads();   // previous hT consumers done
        #pragma unroll
        for (int t = 0; t < 8; ++t) {
            int col = wc0 + t * 16 + (lane & 15);
            float bv = b1[col];
            #pragma unroll
            for (int v = 0; v < 8; ++v) {
                int row = (lane < 16) ? v : v + 8;
                hT[row * HSTR + col] = acc[half][t][v] + bv;
            }
        }
        __syncthreads();

        // Each wave handles 2 rows of this half: LN + ReLU + top-K threshold.
        for (int j = 0; j < 2; ++j) {
            int r = wave * 2 + j;          // 0..15 within the half
            float vals[32];
            float s = 0.f, s2 = 0.f;
            #pragma unroll
            for (int i = 0; i < 32; ++i) {
                float v = hT[r * HSTR + lane + i * 32];
                vals[i] = v; s += v; s2 += v * v;
            }
            #pragma unroll
            for (int m = 16; m >= 1; m >>= 1) {
                s  += __shfl_xor(s,  m, 32);
                s2 += __shfl_xor(s2, m, 32);
            }
            float mean = s * (1.f / HID);
            float var  = s2 * (1.f / HID) - mean * mean;
            float rstd = rsqrtf(var + 1e-5f);

            float mx = 0.f;
            #pragma unroll
            for (int i = 0; i < 32; ++i) {
                int col = lane + i * 32;
                float v = (vals[i] - mean) * rstd * gamma[col] + beta[col];
                v = fmaxf(v, 0.f);
                vals[i] = v;
                mx = fmaxf(mx, v);
            }
            #pragma unroll
            for (int m = 16; m >= 1; m >>= 1) mx = fmaxf(mx, __shfl_xor(mx, m, 32));

            // Bisection for the top-K threshold. Counting uses ballot+popcount:
            // v_cmp feeds s_bcnt on the scalar pipe, count is wave-uniform.
            float lo = 0.f, hi = mx;
            for (int it = 0; it < 20; ++it) {
                float mid = 0.5f * (lo + hi);
                int c = 0;
                #pragma unroll
                for (int i = 0; i < 32; ++i)
                    c += (int)__popcll(__ballot(vals[i] > mid));
                if (c > KTOP) lo = mid; else hi = mid;   // uniform (SALU) branch
            }
            float thr = lo;
            size_t rowg = (size_t)(row0 + half * 16 + r);
            #pragma unroll
            for (int i = 0; i < 32; ++i) {
                float v = vals[i];
                code[rowg * HID + lane + i * 32] = (v > thr) ? v : 0.f;
            }
        }
    }
}

// Decoder: code[32 x 1024] @ W2 -> +b2 -> tanh -> recon
__global__ void __launch_bounds__(256) sae_decoder(const float* __restrict__ code,
                                                   const unsigned short* __restrict__ w2f,
                                                   const float* __restrict__ b2,
                                                   float* __restrict__ recon) {
    extern __shared__ char smem[];
    unsigned short* aF = (unsigned short*)smem;   // 2*KT2*512 bf16 (64 KB)

    const int tid  = threadIdx.x;
    const int lane = tid & 31;
    const int wave = tid >> 5;
    const int row0 = blockIdx.x * ROWS_WG;

    for (int idx = tid; idx < ROWS_WG * HID; idx += 256) {
        int m = idx >> 10;               // 0..31
        int k = idx & 1023;
        float v = code[(size_t)(row0 + m) * HID + k];
        int tile = m >> 4, mm = m & 15;
        int kt = k >> 5, kk = k & 31, l, e;
        a_lane_elem(mm, kk, l, e);
        aF[tile * (KT2 * 512) + kt * 512 + l * 16 + e] = f2bf(v);
    }
    __syncthreads();

    v8f acc[2][6];
    #pragma unroll
    for (int rt = 0; rt < 2; ++rt)
        #pragma unroll
        for (int t = 0; t < 6; ++t)
            #pragma unroll
            for (int v = 0; v < 8; ++v) acc[rt][t][v] = 0.f;

    const int ntBase = wave * 6;           // each wave owns 96 of 768 cols
    for (int kt = 0; kt < KT2; ++kt) {
        v16bf a0 = load_frag(aF + kt * 512 + lane * 16);
        v16bf a1 = load_frag(aF + KT2 * 512 + kt * 512 + lane * 16);
        #pragma unroll
        for (int t = 0; t < 6; ++t) {
            v16bf b = load_frag(w2f + ((size_t)(ntBase + t) * KT2 + kt) * 512 + lane * 16);
            acc[0][t] = __builtin_amdgcn_wmma_f32_16x16x32_bf16(
                false, a0, false, b, (short)0, acc[0][t], false, false);
            acc[1][t] = __builtin_amdgcn_wmma_f32_16x16x32_bf16(
                false, a1, false, b, (short)0, acc[1][t], false, false);
        }
    }

    const int wc0 = wave * 96;
    #pragma unroll
    for (int rt = 0; rt < 2; ++rt) {
        #pragma unroll
        for (int t = 0; t < 6; ++t) {
            int col = wc0 + t * 16 + (lane & 15);
            float bv = b2[col];
            #pragma unroll
            for (int v = 0; v < 8; ++v) {
                int row = (lane < 16) ? v : v + 8;
                recon[(size_t)(row0 + rt * 16 + row) * IN_DIM + col] = fast_tanh(acc[rt][t][v] + bv);
            }
        }
    }
}

extern "C" void kernel_launch(void* const* d_in, const int* in_sizes, int n_in,
                              void* d_out, int out_size, void* d_ws, size_t ws_size,
                              hipStream_t stream) {
    const float* x     = (const float*)d_in[0];
    const float* W1    = (const float*)d_in[1];
    const float* b1    = (const float*)d_in[2];
    const float* gamma = (const float*)d_in[3];
    const float* beta  = (const float*)d_in[4];
    const float* W2    = (const float*)d_in[5];
    const float* b2    = (const float*)d_in[6];

    const int N = in_sizes[0] / IN_DIM;   // 131072

    float* out   = (float*)d_out;
    float* recon = out;                        // [N, 768]
    float* code  = out + (size_t)N * IN_DIM;   // [N, 1024]

    unsigned short* w1f = (unsigned short*)d_ws;              // 1.5 MB bf16 frags
    unsigned short* w2f = w1f + (size_t)IN_DIM * HID;         // 1.5 MB bf16 frags

    // One-time weight conversion into WMMA-fragment layout (L2-resident after).
    prep_w<<<(IN_DIM * HID + 255) / 256, 256, 0, stream>>>(W1, w1f, KT1, HID, IN_DIM * HID);
    prep_w<<<(HID * IN_DIM + 255) / 256, 256, 0, stream>>>(W2, w2f, KT2, IN_DIM, HID * IN_DIM);

    const int nblocks = N / ROWS_WG;
    const size_t sh1 = (size_t)2 * KT1 * 512 * sizeof(unsigned short) + (size_t)16 * HSTR * sizeof(float);
    sae_encoder<<<nblocks, 256, sh1, stream>>>(x, w1f, b1, gamma, beta, code);

    const size_t sh2 = (size_t)2 * KT2 * 512 * sizeof(unsigned short);
    sae_decoder<<<nblocks, 256, sh2, stream>>>(code, w2f, b2, recon);
}
